// BERTSelfAttention_49211735278152
// MI455X (gfx1250) — matrix operand order repaired and checked by reading, not compiled, not run on hardware
//
#include <hip/hip_runtime.h>

typedef __bf16 bf16;
typedef __attribute__((ext_vector_type(4)))  __bf16 v4bf;
typedef __attribute__((ext_vector_type(8)))  __bf16 v8bf;
typedef __attribute__((ext_vector_type(16))) __bf16 v16bf;
typedef __attribute__((ext_vector_type(8)))  float  v8f;
typedef __attribute__((ext_vector_type(4)))  int    i32x4;

__device__ __forceinline__ v8f zero8() { v8f z = {}; return z; }
__device__ __forceinline__ v16bf cat16(v8bf lo, v8bf hi) {
    return __builtin_shufflevector(lo, hi, 0,1,2,3,4,5,6,7,8,9,10,11,12,13,14,15);
}
#define WMMA_BF16(A,B,C) \
    __builtin_amdgcn_wmma_f32_16x16x32_bf16(false,(A),false,(B),(short)0,(C),false,false)

// -------- async global->LDS path (guarded; falls back to VGPR round trip) --------
#if defined(__gfx1250__) && __has_builtin(__builtin_amdgcn_global_load_async_to_lds_b128)
#define USE_ASYNC_LDS 1
#else
#define USE_ASYNC_LDS 0
#endif

#if USE_ASYNC_LDS
typedef __attribute__((address_space(1))) i32x4 gi32x4;   // global (AS1, "__device__")
typedef __attribute__((address_space(3))) i32x4 li32x4;   // LDS    (AS3, "__shared__")
#endif

__device__ __forceinline__ void async_copy16(void* lds, const void* g) {
#if USE_ASYNC_LDS
    __builtin_amdgcn_global_load_async_to_lds_b128((gi32x4*)g, (li32x4*)lds, 0, 0);
#else
    *(v8bf*)lds = *(const v8bf*)g;
#endif
}
__device__ __forceinline__ void wait_async_all() {
#if USE_ASYNC_LDS
#if __has_builtin(__builtin_amdgcn_s_wait_asynccnt)
    __builtin_amdgcn_s_wait_asynccnt(0);
#else
    asm volatile("s_wait_asynccnt 0x0" ::: "memory");
#endif
#endif
}

constexpr int Bc = 4, Sc = 2048, Dc = 1024, Hc = 16, DHc = 64;
constexpr int Mtot = Bc * Sc;            // 8192 rows

// ---------------- fp32 -> bf16 conversion pass (done once, HBM-bound) ----------------
__global__ __launch_bounds__(256)
void to_bf16(const float* __restrict__ src, bf16* __restrict__ dst, int n4) {
    int i = blockIdx.x * blockDim.x + threadIdx.x;
    if (i < n4) {
        float4 v = ((const float4*)src)[i];
        v4bf p = { (bf16)v.x, (bf16)v.y, (bf16)v.z, (bf16)v.w };
        ((v4bf*)dst)[i] = p;
    }
}

// ---------------- QKV projection: out[B,H,S,DH] (bf16) = X @ W^T + b ----------------
constexpr int BM = 128, BN = 128, BK = 64;
constexpr int LDA = BK + 8;              // 72 elems = 144B row stride (16B aligned)

__global__ __launch_bounds__(256, 2)
void qkv_gemm(const bf16* __restrict__ Xb, const bf16* __restrict__ Wb,
              const float* __restrict__ bias, bf16* __restrict__ out, float scale)
{
    __shared__ bf16 As[2][BM * LDA];
    __shared__ bf16 Bs[2][BN * LDA];

    const int tid  = threadIdx.x;
    const int lane = tid & 31;
    const int wave = tid >> 5;
    const int wm0  = (wave & 3) * 32;    // 4 M-waves
    const int wn0  = (wave >> 2) * 64;   // 2 N-waves
    const int l    = lane & 15;
    const int half = lane >> 4;
    const int m0   = blockIdx.y * BM;
    const int n0   = blockIdx.x * BN;

    auto stage = [&](int buf, int k0) {
#pragma unroll
        for (int it = 0; it < 4; ++it) {
            int ci = tid + it * 256;          // 1024 16B chunks per operand tile
            int r  = ci >> 3;
            int c  = (ci & 7) * 8;
            async_copy16(&As[buf][r * LDA + c], &Xb[(size_t)(m0 + r) * Dc + k0 + c]);
            async_copy16(&Bs[buf][r * LDA + c], &Wb[(size_t)(n0 + r) * Dc + k0 + c]);
        }
    };

    v8f acc[2][4];
#pragma unroll
    for (int i = 0; i < 2; ++i)
#pragma unroll
        for (int j = 0; j < 4; ++j) acc[i][j] = zero8();

    stage(0, 0);
    wait_async_all();
    __syncthreads();

    int buf = 0;
    for (int k0 = 0; k0 < Dc; k0 += BK) {
        if (k0 + BK < Dc) stage(buf ^ 1, k0 + BK);   // overlap DMA with WMMA
#pragma unroll
        for (int kc = 0; kc < BK; kc += 32) {
            v16bf a[2], b[4];
#pragma unroll
            for (int wm = 0; wm < 2; ++wm) {
                const bf16* p = &As[buf][(wm0 + wm * 16 + l) * LDA + kc];
                a[wm] = cat16(*(const v8bf*)(p + half * 16),
                              *(const v8bf*)(p + half * 16 + 8));
            }
#pragma unroll
            for (int wn = 0; wn < 4; ++wn) {
                const bf16* p = &Bs[buf][(wn0 + wn * 16 + l) * LDA + kc + half * 16];
                b[wn] = cat16(*(const v8bf*)p, *(const v8bf*)(p + 8));
            }
#pragma unroll
            for (int wm = 0; wm < 2; ++wm)
#pragma unroll
                for (int wn = 0; wn < 4; ++wn)
                    acc[wm][wn] = WMMA_BF16(a[wm], b[wn], acc[wm][wn]);
        }
        wait_async_all();
        __syncthreads();
        buf ^= 1;
    }

    // epilogue: +bias, *scale (1/sqrt(DH) folded into Q), scatter to [B,H,S,DH] bf16
#pragma unroll
    for (int wm = 0; wm < 2; ++wm)
#pragma unroll
        for (int wn = 0; wn < 4; ++wn) {
            int n  = n0 + wn0 + wn * 16 + l;
            float bv = bias[n];
            int h = n >> 6, dh = n & 63;
#pragma unroll
            for (int i = 0; i < 8; ++i) {
                int m = m0 + wm0 + wm * 16 + half * 8 + i;
                int bb = m >> 11, s = m & (Sc - 1);
                float v = (acc[wm][wn][i] + bv) * scale;
                out[(((size_t)(bb * Hc + h)) * Sc + s) * DHc + dh] = (bf16)v;
            }
        }
}

// ---------------- Flash attention: out[B,S,D] (f32) ----------------
constexpr int BQ = 64, BKV = 64;
constexpr int LDR = DHc + 8;             // 72 elems = 144B row stride

__global__ __launch_bounds__(128, 2)
void flash_attn(const bf16* __restrict__ Q, const bf16* __restrict__ K,
                const bf16* __restrict__ V, const float* __restrict__ mask,
                float* __restrict__ out)
{
    __shared__ bf16 Qs[BQ * LDR];
    __shared__ bf16 Ks[2][BKV * LDR];
    __shared__ bf16 Vs[2][DHc * LDR];    // transposed: [dh][key]
    __shared__ bf16 Ps[BQ * LDR];        // [q][key], wave-private rows

    const int tid  = threadIdx.x;
    const int lane = tid & 31;
    const int wave = tid >> 5;
    const int l    = lane & 15;
    const int half = lane >> 4;
    const int q0   = blockIdx.x * BQ;
    const int h    = blockIdx.y;
    const int b    = blockIdx.z;
    const size_t bh = ((size_t)b * Hc + h) * (size_t)Sc * DHc;
    const float* mrow = mask + (size_t)b * Sc;
    const float LOG2E = 1.4426950408889634f;

    auto stage = [&](int buf, int kv0) {
#pragma unroll
        for (int it = 0; it < 4; ++it) {
            int ci = tid + it * 128;         // 512 16B chunks per tile
            int r  = ci >> 3;
            int c  = (ci & 7) * 8;
            async_copy16(&Ks[buf][r * LDR + c], &K[bh + (size_t)(kv0 + r) * DHc + c]);
            v8bf vv = *(const v8bf*)&V[bh + (size_t)(kv0 + r) * DHc + c];
#pragma unroll
            for (int j = 0; j < 8; ++j) Vs[buf][(c + j) * LDR + r] = vv[j];
        }
        // warm L2 for the tile after this one
        if (kv0 + BKV < Sc) {
            const char* pk = (const char*)&K[bh + (size_t)(kv0 + BKV) * DHc];
            const char* pv = (const char*)&V[bh + (size_t)(kv0 + BKV) * DHc];
            __builtin_prefetch(pk + tid * 64, 0, 1);
            __builtin_prefetch(pv + tid * 64, 0, 1);
        }
    };

    // stage Q tile (already scaled by 1/sqrt(DH)) via async path too
#pragma unroll
    for (int it = 0; it < 4; ++it) {
        int ci = tid + it * 128;
        int r  = ci >> 3;
        int c  = (ci & 7) * 8;
        async_copy16(&Qs[r * LDR + c], &Q[bh + (size_t)(q0 + r) * DHc + c]);
    }

    v8f o[4];
#pragma unroll
    for (int i = 0; i < 4; ++i) o[i] = zero8();
    float mrun[8], lrun[8];
#pragma unroll
    for (int i = 0; i < 8; ++i) { mrun[i] = -1e30f; lrun[i] = 0.0f; }

    stage(0, 0);
    wait_async_all();
    __syncthreads();

    int buf = 0;
    for (int kv0 = 0; kv0 < Sc; kv0 += BKV) {
        if (kv0 + BKV < Sc) stage(buf ^ 1, kv0 + BKV);   // overlap next tile with compute

        // S = Q K^T  (16 q-rows per wave x 64 keys)
        v8f sc[4];
#pragma unroll
        for (int i = 0; i < 4; ++i) sc[i] = zero8();
#pragma unroll
        for (int kc = 0; kc < DHc; kc += 32) {
            const bf16* pa = &Qs[(wave * 16 + l) * LDR + kc];
            v16bf af = cat16(*(const v8bf*)(pa + half * 16),
                             *(const v8bf*)(pa + half * 16 + 8));
#pragma unroll
            for (int nt = 0; nt < 4; ++nt) {
                const bf16* pb = &Ks[buf][(nt * 16 + l) * LDR + kc + half * 16];
                v16bf bfr = cat16(*(const v8bf*)pb, *(const v8bf*)(pb + 8));
                sc[nt] = WMMA_BF16(af, bfr, sc[nt]);
            }
        }
        // additive mask (broadcast over q)
#pragma unroll
        for (int nt = 0; nt < 4; ++nt) {
            float mv = mrow[kv0 + nt * 16 + l];
#pragma unroll
            for (int i = 0; i < 8; ++i) sc[nt][i] += mv;
        }

        // online softmax; per-row stats align with the C-layout register mapping
        float pv[4][8];
#pragma unroll
        for (int i = 0; i < 8; ++i) {
            float mx = fmaxf(fmaxf(sc[0][i], sc[1][i]), fmaxf(sc[2][i], sc[3][i]));
#pragma unroll
            for (int off = 1; off < 16; off <<= 1)
                mx = fmaxf(mx, __shfl_xor(mx, off, 32));
            float mnew  = fmaxf(mrun[i], mx);
            float alpha = exp2f((mrun[i] - mnew) * LOG2E);
            float rs = 0.0f;
#pragma unroll
            for (int nt = 0; nt < 4; ++nt) {
                float p = exp2f((sc[nt][i] - mnew) * LOG2E);
                pv[nt][i] = p;
                rs += p;
            }
#pragma unroll
            for (int off = 1; off < 16; off <<= 1)
                rs += __shfl_xor(rs, off, 32);
            lrun[i] = lrun[i] * alpha + rs;
            mrun[i] = mnew;
#pragma unroll
            for (int nt = 0; nt < 4; ++nt) o[nt][i] *= alpha;
        }

        // repack P: C-layout regs -> LDS -> A-layout fragments (wave-private rows)
#pragma unroll
        for (int nt = 0; nt < 4; ++nt)
#pragma unroll
            for (int i = 0; i < 8; ++i)
                Ps[(wave * 16 + half * 8 + i) * LDR + nt * 16 + l] = (bf16)pv[nt][i];
        __syncthreads();

        // O += P V
#pragma unroll
        for (int kc = 0; kc < BKV; kc += 32) {
            const bf16* pa = &Ps[(wave * 16 + l) * LDR + kc];
            v16bf af = cat16(*(const v8bf*)(pa + half * 16),
                             *(const v8bf*)(pa + half * 16 + 8));
#pragma unroll
            for (int nt = 0; nt < 4; ++nt) {
                const bf16* pb = &Vs[buf][(nt * 16 + l) * LDR + kc + half * 16];
                v16bf bfr = cat16(*(const v8bf*)pb, *(const v8bf*)(pb + 8));
                o[nt] = WMMA_BF16(af, bfr, o[nt]);
            }
        }

        wait_async_all();     // next K tile landed
        __syncthreads();      // everyone done reading current buffers
        buf ^= 1;
    }

    // normalize and store f32 [B,S,D]
#pragma unroll
    for (int i = 0; i < 8; ++i) {
        float inv = 1.0f / lrun[i];
        int q = q0 + wave * 16 + half * 8 + i;
        size_t rowbase = ((size_t)b * Sc + q) * Dc + (size_t)h * DHc;
#pragma unroll
        for (int nt = 0; nt < 4; ++nt)
            out[rowbase + nt * 16 + l] = o[nt][i] * inv;
    }
}

extern "C" void kernel_launch(void* const* d_in, const int* in_sizes, int n_in,
                              void* d_out, int out_size, void* d_ws, size_t ws_size,
                              hipStream_t stream) {
    const float* X    = (const float*)d_in[0];
    const float* mask = (const float*)d_in[1];
    const float* Wq   = (const float*)d_in[2];
    const float* bq   = (const float*)d_in[3];
    const float* Wk   = (const float*)d_in[4];
    const float* bk   = (const float*)d_in[5];
    const float* Wv   = (const float*)d_in[6];
    const float* bv   = (const float*)d_in[7];
    float* out = (float*)d_out;

    const size_t MX = (size_t)Mtot * Dc;   // 8,388,608 elems
    const size_t WN = (size_t)Dc * Dc;     // 1,048,576 elems
    bf16* Xb  = (bf16*)d_ws;
    bf16* Wqb = Xb  + MX;
    bf16* Wkb = Wqb + WN;
    bf16* Wvb = Wkb + WN;
    bf16* Qb  = Wvb + WN;
    bf16* Kb  = Qb  + MX;
    bf16* Vb  = Kb  + MX;

    // one-shot fp32 -> bf16 (HBM-bound, ~46 MB read)
    to_bf16<<<(int)(MX / 4 / 256), 256, 0, stream>>>(X,  Xb,  (int)(MX / 4));
    to_bf16<<<(int)(WN / 4 / 256), 256, 0, stream>>>(Wq, Wqb, (int)(WN / 4));
    to_bf16<<<(int)(WN / 4 / 256), 256, 0, stream>>>(Wk, Wkb, (int)(WN / 4));
    to_bf16<<<(int)(WN / 4 / 256), 256, 0, stream>>>(Wv, Wvb, (int)(WN / 4));

    dim3 gg(Dc / BN, Mtot / BM, 1);
    qkv_gemm<<<gg, 256, 0, stream>>>(Xb, Wqb, bq, Qb, 0.125f);  // 1/sqrt(64) folded into Q
    qkv_gemm<<<gg, 256, 0, stream>>>(Xb, Wkb, bk, Kb, 1.0f);
    qkv_gemm<<<gg, 256, 0, stream>>>(Xb, Wvb, bv, Vb, 1.0f);

    flash_attn<<<dim3(Sc / BQ, Hc, Bc), 128, 0, stream>>>(Qb, Kb, Vb, mask, out);
}
